// GCNPairTwoConv_21500606284485
// MI455X (gfx1250) — matrix operand order repaired and checked by reading, not compiled, not run on hardware
//
#include <hip/hip_runtime.h>
#include <hip/hip_bf16.h>
#include <math.h>

typedef __attribute__((ext_vector_type(16))) _Float16 v16h;
typedef __attribute__((ext_vector_type(8)))  _Float16 v8h;
typedef __attribute__((ext_vector_type(8)))  float    v8f;

#define NNODE 50000
#define NEDGE 800000
#define FINF  64
#define HID   96
#define NG    256

#define CDIV(a, b) (((a) + (b) - 1) / (b))

// ---------- helpers: monotonic float<->uint encoding for atomicMax ----------
__device__ __forceinline__ unsigned enc_f(float f) {
  unsigned u = __float_as_uint(f);
  return (u & 0x80000000u) ? ~u : (u | 0x80000000u);
}
__device__ __forceinline__ float dec_f(unsigned u) {
  return (u & 0x80000000u) ? __uint_as_float(u ^ 0x80000000u) : __uint_as_float(~u);
}

// ---------- f32 -> f16 convert ----------
__global__ void k_f32_to_f16(const float* __restrict__ in, _Float16* __restrict__ out, int n) {
  int i = blockIdx.x * blockDim.x + threadIdx.x;
  if (i < n) out[i] = (_Float16)in[i];
}

// ---------- pack W[K x Nout] (f32 row-major) into WMMA B-fragment layout (f16) ----------
// packed index: ((kt*colTiles + ct)*32 + lane)*16 + j
//   K_ = kt*32 + (lane>>4)*16 + j ; N_ = ct*16 + (lane&15)
__global__ void k_pack_w(const float* __restrict__ W, _Float16* __restrict__ out, int K, int Nout) {
  int idx = blockIdx.x * blockDim.x + threadIdx.x;
  int total = K * Nout;
  if (idx >= total) return;
  int colTiles = Nout >> 4;
  int per_kt = colTiles * 512;
  int kt   = idx / per_kt;
  int rem  = idx % per_kt;
  int ct   = rem / 512;
  int lane = (rem % 512) / 16;
  int j    = rem % 16;
  int Kidx = kt * 32 + (lane >> 4) * 16 + j;
  int Nidx = ct * 16 + (lane & 15);
  out[idx] = (_Float16)W[Kidx * Nout + Nidx];
}

// ---------- WMMA GEMM with fused GCN self-loop epilogue ----------
// C[M x Nout] = relu?(A[M x K] @ B + bias)
// if dinv != null:  also  agg[m, n] = dinv[m]^2 * C[m, n] + b2[n]
// A f16 row-major, Bp packed fragments, C f32 row-major.
// grid.x = ceil(M/16); block = (32, Nout/16); one wave per 16x16 C tile.
__global__ void k_wmma_gemm(const _Float16* __restrict__ A,
                            const _Float16* __restrict__ Bp,
                            const float* __restrict__ bias,
                            float* __restrict__ C,
                            const float* __restrict__ dinv,
                            const float* __restrict__ b2,
                            float* __restrict__ agg,
                            int M, int K, int Nout, int relu) {
  const int lane = threadIdx.x;      // 0..31
  const int ct   = threadIdx.y;      // column tile
  const int colTiles = Nout >> 4;
  const int half = lane >> 4;
  const int r    = lane & 15;
  const int rowTile = blockIdx.x;
  int row = rowTile * 16 + r;
  if (row >= M) row = M - 1;         // clamp (EXEC stays all-ones for WMMA)
  const int col = ct * 16 + r;

  v8f c;
  const float binit = bias ? bias[col] : 0.0f;
#pragma unroll
  for (int i = 0; i < 8; ++i) c[i] = binit;

  const _Float16* arow = A + (size_t)row * K;
  const int kTiles = K >> 5;
  for (int kt = 0; kt < kTiles; ++kt) {
    const int k = kt << 5;
    // A frag: lanes 0-15 hold K = {k..k+15}; lanes 16-31 hold {k+16..k+31}
    v8h alo = *(const v8h*)(arow + k + half * 16);
    v8h ahi = *(const v8h*)(arow + k + half * 16 + 8);
    v16h a;
#pragma unroll
    for (int j = 0; j < 8; ++j) { a[j] = alo[j]; a[j + 8] = ahi[j]; }
    v16h b = *(const v16h*)(Bp + ((size_t)(kt * colTiles + ct) * 32 + lane) * 16);
    c = __builtin_amdgcn_wmma_f32_16x16x32_f16(false, a, false, b, (short)0, c, false, false);
  }

  const float b2v = dinv ? b2[col] : 0.0f;
#pragma unroll
  for (int i = 0; i < 8; ++i) {
    int orow = rowTile * 16 + i + 8 * half;   // C layout: VGPR i -> row i + 8*half
    if (orow < M) {
      float v = c[i];
      if (relu) v = fmaxf(v, 0.0f);
      C[(size_t)orow * Nout + col] = v;
      if (dinv) {
        float di = dinv[orow];
        agg[(size_t)orow * Nout + col] = di * di * v + b2v;
      }
    }
  }
}

// ---------- attention gate: gate[n] = gb2 + dot(ghid[n,:64], gW2); track segment max ----------
__global__ void k_gate(const float* __restrict__ ghid, const float* __restrict__ gW2,
                       const float* __restrict__ gb2, const int* __restrict__ batch,
                       float* __restrict__ gate, unsigned* __restrict__ mkey, int n) {
  int i = blockIdx.x * blockDim.x + threadIdx.x;
  if (i >= n) return;
  const float* row = ghid + (size_t)i * FINF;
  float acc = gb2[0];
#pragma unroll 8
  for (int f = 0; f < FINF; ++f) acc += row[f] * gW2[f];
  gate[i] = acc;
  atomicMax(&mkey[batch[i]], enc_f(acc));
}

__global__ void k_gate_exp(float* __restrict__ gate, const int* __restrict__ batch,
                           const unsigned* __restrict__ mkey, float* __restrict__ denom, int n) {
  int i = blockIdx.x * blockDim.x + threadIdx.x;
  if (i >= n) return;
  int g = batch[i];
  float e = expf(gate[i] - dec_f(mkey[g]));
  gate[i] = e;
  atomicAdd(&denom[g], e);
}

// thread per (node, 4-feature chunk): b128 read of x, 4 atomic adds
__global__ void k_att_scatter4(const float* __restrict__ gate, const float* __restrict__ denom,
                               const int* __restrict__ batch, const float* __restrict__ x,
                               float* __restrict__ att, int n) {
  const int CH = FINF / 4;   // 16
  int idx = blockIdx.x * blockDim.x + threadIdx.x;
  if (idx >= n * CH) return;
  int i = idx / CH, c = idx % CH;
  int g = batch[i];
  float alpha = gate[i] / denom[g];
  const float4 v = *(const float4*)(x + (size_t)i * FINF + c * 4);
  float* o = att + g * FINF + c * 4;
  atomicAdd(o + 0, alpha * v.x);
  atomicAdd(o + 1, alpha * v.y);
  atomicAdd(o + 2, alpha * v.z);
  atomicAdd(o + 3, alpha * v.w);
}

// ---------- degree / dinv ----------
__global__ void k_deg(const int* __restrict__ ei, float* __restrict__ deg, int e) {
  int i = blockIdx.x * blockDim.x + threadIdx.x;
  if (i < e) atomicAdd(&deg[ei[NEDGE + i]], 1.0f);
}
__global__ void k_dinv(float* __restrict__ d, int n) {
  int i = blockIdx.x * blockDim.x + threadIdx.x;
  if (i < n) d[i] = rsqrtf(d[i] + 1.0f);
}

// ---------- edge scatter: agg[dst] += dinv[src]*dinv[dst] * xw[src] ----------
// thread per (edge, 4-feature chunk): b128 read of xw, 4 atomic adds
__global__ void k_scatter4(const int* __restrict__ ei, const float* __restrict__ dinv,
                           const float* __restrict__ xw, float* __restrict__ agg, int e) {
  const int CH = HID / 4;    // 24
  int idx = blockIdx.x * blockDim.x + threadIdx.x;
  if (idx >= e * CH) return;
  int ed = idx / CH, c = idx % CH;
  int src = ei[ed], dst = ei[NEDGE + ed];
  float norm = dinv[src] * dinv[dst];
  const float4 v = *(const float4*)(xw + (size_t)src * HID + c * 4);
  float* o = agg + (size_t)dst * HID + c * 4;
  atomicAdd(o + 0, norm * v.x);
  atomicAdd(o + 1, norm * v.y);
  atomicAdd(o + 2, norm * v.z);
  atomicAdd(o + 3, norm * v.w);
}

// ---------- relu + f16 convert for next layer ----------
__global__ void k_relu_f16(const float* __restrict__ in, _Float16* __restrict__ out, int n) {
  int i = blockIdx.x * blockDim.x + threadIdx.x;
  if (i < n) out[i] = (_Float16)fmaxf(in[i], 0.0f);
}

// ---------- mean pool ----------
__global__ void k_cnt(const int* __restrict__ batch, float* __restrict__ cnt, int n) {
  int i = blockIdx.x * blockDim.x + threadIdx.x;
  if (i < n) atomicAdd(&cnt[batch[i]], 1.0f);
}
// thread per (node, 4-feature chunk)
__global__ void k_psum4(const float* __restrict__ agg, const int* __restrict__ batch,
                        float* __restrict__ psum, int n) {
  const int CH = HID / 4;    // 24
  int idx = blockIdx.x * blockDim.x + threadIdx.x;
  if (idx >= n * CH) return;
  int i = idx / CH, c = idx % CH;
  int g = batch[i];
  const float4 v = *(const float4*)(agg + (size_t)i * HID + c * 4);
  float* o = psum + g * HID + c * 4;
  atomicAdd(o + 0, v.x);
  atomicAdd(o + 1, v.y);
  atomicAdd(o + 2, v.z);
  atomicAdd(o + 3, v.w);
}
__global__ void k_pooldiv(const float* __restrict__ psum, const float* __restrict__ cnt,
                          float* __restrict__ pool) {
  int idx = blockIdx.x * blockDim.x + threadIdx.x;
  if (idx >= NG * HID) return;
  int g = idx / HID;
  pool[idx] = psum[idx] / fmaxf(cnt[g], 1.0f);
}

// ---------- final MLP ----------
__global__ void k_fin(const float* __restrict__ pp, const float* __restrict__ pd,
                      const float* __restrict__ ap, const float* __restrict__ ad,
                      const float* __restrict__ lW0, const float* __restrict__ lb0,
                      float* __restrict__ fin) {
  int g = blockIdx.x, f = threadIdx.x;   // grid NG, block HID
  float acc = lb0[f];
  for (int j = 0; j < HID; ++j)  acc += pp[g * HID + j] * lW0[j * HID + f];
  for (int j = 0; j < HID; ++j)  acc += pd[g * HID + j] * lW0[(HID + j) * HID + f];
  for (int j = 0; j < FINF; ++j) acc += ap[g * FINF + j] * lW0[(2 * HID + j) * HID + f];
  for (int j = 0; j < FINF; ++j) acc += ad[g * FINF + j] * lW0[(2 * HID + FINF + j) * HID + f];
  fin[g * HID + f] = fmaxf(acc, 0.0f);
}
__global__ void k_out(const float* __restrict__ fin, const float* __restrict__ lW1,
                      const float* __restrict__ lb1, float* __restrict__ out) {
  int g = threadIdx.x;                   // block NG
  float acc = lb1[0];
  for (int f = 0; f < HID; ++f) acc += fin[g * HID + f] * lW1[f];
  out[g] = acc;
}

extern "C" void kernel_launch(void* const* d_in, const int* in_sizes, int n_in,
                              void* d_out, int out_size, void* d_ws, size_t ws_size,
                              hipStream_t stream) {
  (void)in_sizes; (void)n_in; (void)out_size; (void)ws_size;

  const float* x_p  = (const float*)d_in[0];
  const float* x_d  = (const float*)d_in[1];
  const int*   ei_p = (const int*)d_in[4];
  const int*   ei_d = (const int*)d_in[5];
  const int*   bt_p = (const int*)d_in[6];
  const int*   bt_d = (const int*)d_in[7];
  const float* Wp0 = (const float*)d_in[8],  *bp0 = (const float*)d_in[9];
  const float* Wp1 = (const float*)d_in[10], *bp1 = (const float*)d_in[11];
  const float* Wp2 = (const float*)d_in[12], *bp2 = (const float*)d_in[13];
  const float* Wd0 = (const float*)d_in[14], *bd0 = (const float*)d_in[15];
  const float* Wd1 = (const float*)d_in[16], *bd1 = (const float*)d_in[17];
  const float* Wd2 = (const float*)d_in[18], *bd2 = (const float*)d_in[19];
  const float* gW1 = (const float*)d_in[20], *gb1 = (const float*)d_in[21];
  const float* gW2 = (const float*)d_in[22], *gb2 = (const float*)d_in[23];
  const float* lW0 = (const float*)d_in[24], *lb0 = (const float*)d_in[25];
  const float* lW1 = (const float*)d_in[26], *lb1 = (const float*)d_in[27];

  // ---- workspace layout (256B-aligned cursor) ----
  char* base = (char*)d_ws;
  size_t off = 0;
  auto alloc = [&](size_t bytes) -> void* {
    void* p = base + off;
    off = (off + bytes + 255) & ~(size_t)255;
    return p;
  };
  // packed f16 weights
  _Float16* wp0_16 = (_Float16*)alloc(FINF * HID * 2);
  _Float16* wp1_16 = (_Float16*)alloc(HID * HID * 2);
  _Float16* wp2_16 = (_Float16*)alloc(HID * HID * 2);
  _Float16* wd0_16 = (_Float16*)alloc(FINF * HID * 2);
  _Float16* wd1_16 = (_Float16*)alloc(HID * HID * 2);
  _Float16* wd2_16 = (_Float16*)alloc(HID * HID * 2);
  _Float16* gw1_16 = (_Float16*)alloc(FINF * FINF * 2);
  // per-graph persistent
  float*    att_p  = (float*)alloc(NG * FINF * 4);
  float*    att_d  = (float*)alloc(NG * FINF * 4);
  float*    pool_p = (float*)alloc(NG * HID * 4);
  float*    pool_d = (float*)alloc(NG * HID * 4);
  float*    fin    = (float*)alloc(NG * HID * 4);
  unsigned* mkey   = (unsigned*)alloc(NG * 4);
  float*    denom  = (float*)alloc(NG * 4);
  float*    cnt    = (float*)alloc(NG * 4);
  float*    psum   = (float*)alloc(NG * HID * 4);
  // per-side reusable big scratch
  _Float16* x16  = (_Float16*)alloc((size_t)NNODE * FINF * 2);
  _Float16* h16  = (_Float16*)alloc((size_t)NNODE * HID * 2);
  float*    xw   = (float*)alloc((size_t)NNODE * HID * 4);   // also gate-hidden [N x 64]
  float*    agg  = (float*)alloc((size_t)NNODE * HID * 4);
  float*    dinv = (float*)alloc((size_t)NNODE * 4);         // deg then dinv in-place
  float*    gate = (float*)alloc((size_t)NNODE * 4);

  // ---- pack weights to f16 WMMA B-fragment layout ----
  k_pack_w<<<CDIV(FINF * HID, 256), 256, 0, stream>>>(Wp0, wp0_16, FINF, HID);
  k_pack_w<<<CDIV(HID * HID, 256), 256, 0, stream>>>(Wp1, wp1_16, HID, HID);
  k_pack_w<<<CDIV(HID * HID, 256), 256, 0, stream>>>(Wp2, wp2_16, HID, HID);
  k_pack_w<<<CDIV(FINF * HID, 256), 256, 0, stream>>>(Wd0, wd0_16, FINF, HID);
  k_pack_w<<<CDIV(HID * HID, 256), 256, 0, stream>>>(Wd1, wd1_16, HID, HID);
  k_pack_w<<<CDIV(HID * HID, 256), 256, 0, stream>>>(Wd2, wd2_16, HID, HID);
  k_pack_w<<<CDIV(FINF * FINF, 256), 256, 0, stream>>>(gW1, gw1_16, FINF, FINF);

  const int rowTiles = CDIV(NNODE, 16);
  const dim3 gblk64(32, FINF / 16);   // (32,4)
  const dim3 gblk96(32, HID / 16);    // (32,6)

  auto run_side = [&](const float* x, const int* ei, const int* bt,
                      const _Float16* w0, const float* b0,
                      const _Float16* w1, const float* b1,
                      const _Float16* w2, const float* b2,
                      float* att, float* pool) {
    // zero accumulators
    hipMemsetAsync(dinv, 0, (size_t)NNODE * 4, stream);
    hipMemsetAsync(mkey, 0, NG * 4, stream);           // enc key 0 == below all floats
    hipMemsetAsync(denom, 0, NG * 4, stream);
    hipMemsetAsync(cnt, 0, NG * 4, stream);
    hipMemsetAsync(psum, 0, NG * HID * 4, stream);
    hipMemsetAsync(att, 0, NG * FINF * 4, stream);

    // degree -> dinv (needed by fused GEMM epilogue)
    k_deg<<<CDIV(NEDGE, 256), 256, 0, stream>>>(ei, dinv, NEDGE);
    k_dinv<<<CDIV(NNODE, 256), 256, 0, stream>>>(dinv, NNODE);

    // convert raw features to f16 (feeds both gate GEMM and GCN layer 0)
    k_f32_to_f16<<<CDIV(NNODE * FINF, 256), 256, 0, stream>>>(x, x16, NNODE * FINF);

    // --- attention pool on raw features ---
    k_wmma_gemm<<<rowTiles, gblk64, 0, stream>>>(x16, gw1_16, gb1, xw,
                                                 nullptr, nullptr, nullptr,
                                                 NNODE, FINF, FINF, 1);
    k_gate<<<CDIV(NNODE, 256), 256, 0, stream>>>(xw, gW2, gb2, bt, gate, mkey, NNODE);
    k_gate_exp<<<CDIV(NNODE, 256), 256, 0, stream>>>(gate, bt, mkey, denom, NNODE);
    k_att_scatter4<<<CDIV(NNODE * (FINF / 4), 256), 256, 0, stream>>>(gate, denom, bt, x, att, NNODE);

    // --- GCN layer 0 (F_IN -> H), relu ---  (GEMM writes xw and agg = dinv^2*xw + b)
    k_wmma_gemm<<<rowTiles, gblk96, 0, stream>>>(x16, w0, nullptr, xw, dinv, b0, agg,
                                                 NNODE, FINF, HID, 0);
    k_scatter4<<<CDIV(NEDGE * (HID / 4), 256), 256, 0, stream>>>(ei, dinv, xw, agg, NEDGE);
    k_relu_f16<<<CDIV(NNODE * HID, 256), 256, 0, stream>>>(agg, h16, NNODE * HID);

    // --- GCN layer 1 (H -> H), relu ---
    k_wmma_gemm<<<rowTiles, gblk96, 0, stream>>>(h16, w1, nullptr, xw, dinv, b1, agg,
                                                 NNODE, HID, HID, 0);
    k_scatter4<<<CDIV(NEDGE * (HID / 4), 256), 256, 0, stream>>>(ei, dinv, xw, agg, NEDGE);
    k_relu_f16<<<CDIV(NNODE * HID, 256), 256, 0, stream>>>(agg, h16, NNODE * HID);

    // --- GCN layer 2 (H -> H), no relu ---
    k_wmma_gemm<<<rowTiles, gblk96, 0, stream>>>(h16, w2, nullptr, xw, dinv, b2, agg,
                                                 NNODE, HID, HID, 0);
    k_scatter4<<<CDIV(NEDGE * (HID / 4), 256), 256, 0, stream>>>(ei, dinv, xw, agg, NEDGE);

    // --- mean pool ---
    k_cnt<<<CDIV(NNODE, 256), 256, 0, stream>>>(bt, cnt, NNODE);
    k_psum4<<<CDIV(NNODE * (HID / 4), 256), 256, 0, stream>>>(agg, bt, psum, NNODE);
    k_pooldiv<<<CDIV(NG * HID, 256), 256, 0, stream>>>(psum, cnt, pool);
  };

  run_side(x_p, ei_p, bt_p, wp0_16, bp0, wp1_16, bp1, wp2_16, bp2, att_p, pool_p);
  run_side(x_d, ei_d, bt_d, wd0_16, bd0, wd1_16, bd1, wd2_16, bd2, att_d, pool_d);

  // --- final MLP: [G, 2H+2F] -> H (relu) -> 1 ---
  k_fin<<<NG, HID, 0, stream>>>(pool_p, pool_d, att_p, att_d, lW0, lb0, fin);
  k_out<<<1, NG, 0, stream>>>(fin, lW1, lb1, (float*)d_out);
}
